// TRNN_14413910245609
// MI455X (gfx1250) — compile-verified, matrix operand order checked by reading, not run on hardware
//
#include <hip/hip_runtime.h>
#include <math.h>

// Problem dims (fixed by the reference)
#define S_DIM 1024   // states_dim
#define B_DIM 128    // batch
#define O_DIM 256    // output_dim
#define T_DIM 512    // seq_len

// CDNA5 WMMA vector types (probe-confirmed signatures)
typedef __attribute__((ext_vector_type(16))) __bf16 v16bf;
typedef __attribute__((ext_vector_type(8)))  float  v8f;

union Frag {
  v16bf v;
  uint4 q[2];
  unsigned int u[8];
};

// LDS A-tile: 16 rows x 512 dwords (1024 bf16), stride 516 keeps rows
// 16-byte aligned and spreads rows across banks (516 mod 64 = 4).
#define SH_STRIDE 516

__device__ __forceinline__ unsigned short f32_to_bf16_rne(float f) {
  unsigned int u = __float_as_uint(f);
  u += 0x7FFFu + ((u >> 16) & 1u);
  return (unsigned short)(u >> 16);
}

// ---------------------------------------------------------------------------
// Prep: convert fp32 weights [N,S] row-major into bf16 WMMA B-fragment layout.
// Fragment (nTile,kTile) = 32 lanes x 8 dwords, matching v_wmma_*_16x16x32_bf16
// B-operand striping: lane l -> column n = nTile*16 + (l&15); kb = (l<16)?0:8;
//   dword d<4 :  packs W[n][k0],W[n][k0+1] with k0 = kTile*32 + kb + 2*d
//   dword d>=4:  k0 = kTile*32 + kb + 16 + 2*(d-4)
// ---------------------------------------------------------------------------
__global__ void prep_weights_kernel(const float* __restrict__ w,
                                    unsigned int* __restrict__ wb,
                                    int nTiles, int kTiles) {
  int idx = blockIdx.x * blockDim.x + threadIdx.x;
  int total = nTiles * kTiles * 32 * 8;
  if (idx >= total) return;
  int d    = idx & 7;
  int lane = (idx >> 3) & 31;
  int kT   = (idx >> 8) % kTiles;
  int nT   = (idx >> 8) / kTiles;
  int n    = nT * 16 + (lane & 15);
  int kb   = (lane < 16) ? 0 : 8;
  int k0   = kT * 32 + ((d < 4) ? (kb + 2 * d) : (kb + 16 + 2 * (d - 4)));
  float lo = w[(size_t)n * S_DIM + k0];
  float hi = w[(size_t)n * S_DIM + k0 + 1];
  wb[idx] = (unsigned)f32_to_bf16_rne(lo) | ((unsigned)f32_to_bf16_rne(hi) << 16);
}

// Prep: h workspace copy (inputs are read-only) + tanh(h_init) as bf16 row-major
__global__ void prep_state_kernel(const float* __restrict__ h_init,
                                  float* __restrict__ h_cur,
                                  unsigned short* __restrict__ th0) {
  int i = blockIdx.x * blockDim.x + threadIdx.x;
  if (i >= B_DIM * S_DIM) return;
  float h  = h_init[i];
  h_cur[i] = h;
  th0[i]   = f32_to_bf16_rne(tanhf(h));
}

// ---------------------------------------------------------------------------
// Mainloop: one wave computes a 16x16 f32 tile of  tanh(H) @ W^T  over K=1024.
// The full 16x1024 bf16 A-tile (32 KB) is staged into LDS ONCE (single
// barrier), then 32 WMMA K-steps run with no synchronization so the scheduler
// can pipeline ds_load_b128 / global_load_b128 against v_wmma freely.
// ---------------------------------------------------------------------------
__device__ __forceinline__ v8f gemm_rowtile(const unsigned short* __restrict__ th,
                                            const unsigned int* __restrict__ wfrag,
                                            int row0, int nTile,
                                            unsigned int* sh) {
  const int tid  = threadIdx.x;
  const int lane = tid & 31;
  const uint4* thQ = (const uint4*)th;     // 8 bf16 per uint4
  const uint4* wq  = (const uint4*)wfrag;
  const int kTiles = S_DIM / 32;

  // Stage whole A tile: 16 rows x 128 uint4 = 2048 uint4, 8 per thread.
#pragma unroll
  for (int j = 0; j < 8; ++j) {
    int idx = tid + j * 256;
    int r   = idx >> 7;          // 128 uint4 per row
    int c4  = idx & 127;
    *(uint4*)&sh[r * SH_STRIDE + c4 * 4] = thQ[(size_t)(row0 + r) * 128 + c4];
  }
  __syncthreads();

  v8f acc = {};
#pragma unroll 4
  for (int kT = 0; kT < kTiles; ++kT) {
    Frag a, b;
    // A fragment per ISA layout: lane -> M = lane&15; lanes>=16 take K+8
    int m  = lane & 15;
    int hs = lane >> 4;
    int bu = m * SH_STRIDE + kT * 16 + hs * 4;
    a.q[0] = *(const uint4*)&sh[bu];       // K = kb .. kb+7
    a.q[1] = *(const uint4*)&sh[bu + 8];   // K = kb+16 .. kb+23
    // B fragment: pre-swizzled, 8 consecutive dwords per lane
    size_t f = ((size_t)(nTile * kTiles + kT) * 32 + lane) * 2;
    b.q[0] = wq[f];
    b.q[1] = wq[f + 1];
    acc = __builtin_amdgcn_wmma_f32_16x16x32_bf16(false, a.v, false, b.v,
                                                  (short)0, acc, false, false);
  }
  return acc;
}

// ---------------------------------------------------------------------------
// Fused per-step kernel, 80 blocks:
//   blocks 0..15  : output GEMM for the PREVIOUS step (independent of this
//                   step's state update -> runs in its shadow, off the
//                   critical path). errors[t-1] = th_cur@W_o^T + b_o - x[t-1]
//   blocks 16..79 : state update  h <- 0.75*h + 0.25*(th_cur@W_r^T + b_r),
//                   then th_nxt = bf16(tanh(h_new))
// Both roles only READ th_cur; state writes the other ping-pong buffer.
// ---------------------------------------------------------------------------
__global__ __launch_bounds__(256) void rnn_step_kernel(
    const unsigned short* __restrict__ th_cur,
    unsigned short* __restrict__ th_nxt,
    float* __restrict__ h_cur,
    const unsigned int* __restrict__ w_rb,
    const float* __restrict__ b_r,
    const unsigned int* __restrict__ w_ob,
    const float* __restrict__ b_o,
    const float* __restrict__ x_prev,    // x[t-1]      (unused when do_out==0)
    float* __restrict__ err_prev,        // errors[t-1] (unused when do_out==0)
    int do_out) {
  __shared__ unsigned int sh[16 * SH_STRIDE];
  const int lane = threadIdx.x & 31;
  const int wave = threadIdx.x >> 5;

  if (blockIdx.x < 16) {
    // ---- output role: errors[t-1] ----
    if (!do_out) return;
    const int ob    = blockIdx.x;
    const int row0  = (ob >> 1) * 16;            // 8 row tiles
    const int nTile = (ob & 1) * 8 + wave;       // 16 col tiles (O=256)
    v8f acc = gemm_rowtile(th_cur, w_ob, row0, nTile, sh);
    const int col = nTile * 16 + (lane & 15);
    const float bo = b_o[col];
#pragma unroll
    for (int r = 0; r < 8; ++r) {
      int row = row0 + r + ((lane >> 4) << 3);
      size_t off = (size_t)row * O_DIM + col;
      err_prev[off] = acc[r] + bo - x_prev[off];
    }
  } else {
    // ---- state role ----
    const int sb    = blockIdx.x - 16;
    const int row0  = (sb >> 3) * 16;            // 8 row tiles
    const int nTile = (sb & 7) * 8 + wave;       // 64 col tiles (S=1024)
    v8f acc = gemm_rowtile(th_cur, w_rb, row0, nTile, sh);
    const int col = nTile * 16 + (lane & 15);
    const float br = b_r[col];
#pragma unroll
    for (int r = 0; r < 8; ++r) {
      int row = row0 + r + ((lane >> 4) << 3);   // C layout: VGPR r -> M=r / r+8
      size_t off = (size_t)row * S_DIM + col;
      float hn = 0.75f * h_cur[off] + 0.25f * (acc[r] + br);
      h_cur[off]  = hn;
      th_nxt[off] = f32_to_bf16_rne(tanhf(hn));
    }
  }
}

// Tail: errors[T-1] only (grid (2,8))
__global__ __launch_bounds__(256) void rnn_out_kernel(
    const unsigned short* __restrict__ th,
    const unsigned int* __restrict__ w_ob,
    const float* __restrict__ b_o,
    const float* __restrict__ x_t,
    float* __restrict__ err_t) {
  __shared__ unsigned int sh[16 * SH_STRIDE];
  const int lane  = threadIdx.x & 31;
  const int wave  = threadIdx.x >> 5;
  const int row0  = blockIdx.y * 16;
  const int nTile = blockIdx.x * 8 + wave;

  v8f acc = gemm_rowtile(th, w_ob, row0, nTile, sh);

  const int col = nTile * 16 + (lane & 15);
  const float bo = b_o[col];
#pragma unroll
  for (int r = 0; r < 8; ++r) {
    int row = row0 + r + ((lane >> 4) << 3);
    size_t off = (size_t)row * O_DIM + col;
    err_t[off] = acc[r] + bo - x_t[off];
  }
}

// ---------------------------------------------------------------------------
// Workspace layout (needs ~3.5 MB):
//   [0)            w_rb : S*S bf16 fragments   = 2 MiB
//   [2 MiB)        w_ob : O*S bf16 fragments   = 512 KiB
//   [2.5 MiB)      h    : B*S f32              = 512 KiB
//   [3 MiB)        th0  : B*S bf16             = 256 KiB
//   [3.25 MiB)     th1  : B*S bf16             = 256 KiB
// ---------------------------------------------------------------------------
extern "C" void kernel_launch(void* const* d_in, const int* in_sizes, int n_in,
                              void* d_out, int out_size, void* d_ws, size_t ws_size,
                              hipStream_t stream) {
  const float* x      = (const float*)d_in[0];
  const float* h_init = (const float*)d_in[1];
  const float* w_r    = (const float*)d_in[2];
  const float* b_r    = (const float*)d_in[3];
  const float* w_o    = (const float*)d_in[4];
  const float* b_o    = (const float*)d_in[5];
  float* out = (float*)d_out;

  char* ws = (char*)d_ws;
  unsigned int*   w_rb = (unsigned int*)(ws);
  unsigned int*   w_ob = (unsigned int*)(ws + (2u << 20));
  float*          h    = (float*)(ws + (2u << 20) + (512u << 10));
  unsigned short* th0  = (unsigned short*)(ws + (3u << 20));
  unsigned short* th1  = (unsigned short*)(ws + (3u << 20) + (256u << 10));

  // One-time (per launch) weight swizzle + state init — deterministic.
  {
    int totR = (S_DIM / 16) * (S_DIM / 32) * 256;   // 524288
    int totO = (O_DIM / 16) * (S_DIM / 32) * 256;   // 131072
    prep_weights_kernel<<<(totR + 255) / 256, 256, 0, stream>>>(w_r, w_rb, S_DIM / 16, S_DIM / 32);
    prep_weights_kernel<<<(totO + 255) / 256, 256, 0, stream>>>(w_o, w_ob, O_DIM / 16, S_DIM / 32);
    prep_state_kernel<<<(B_DIM * S_DIM + 255) / 256, 256, 0, stream>>>(h_init, h, th0);
  }

  unsigned short* thp[2] = {th0, th1};
  const size_t BO = (size_t)B_DIM * O_DIM;

  // One fused kernel per timestep: state(t) + output(t-1) hidden behind it.
  for (int t = 0; t < T_DIM; ++t) {
    unsigned short* th_cur = thp[t & 1];
    unsigned short* th_nxt = thp[(t + 1) & 1];
    const float* xp = (t > 0) ? (x   + (size_t)(t - 1) * BO) : x;
    float*       ep = (t > 0) ? (out + (size_t)(t - 1) * BO) : out;
    rnn_step_kernel<<<80, 256, 0, stream>>>(th_cur, th_nxt, h, w_rb, b_r,
                                            w_ob, b_o, xp, ep, (t > 0) ? 1 : 0);
  }
  // Tail: errors[T-1] from th[(T)&1] = th0
  rnn_out_kernel<<<dim3(O_DIM / 128, B_DIM / 16), 256, 0, stream>>>(
      thp[T_DIM & 1], w_ob, b_o,
      x + (size_t)(T_DIM - 1) * BO, out + (size_t)(T_DIM - 1) * BO);
}
